// LePEAttention_12481174962257
// MI455X (gfx1250) — matrix operand order repaired and checked by reading, not compiled, not run on hardware
//
#include <hip/hip_runtime.h>

// ---------------------------------------------------------------------------
// LePE attention (idx=0 branch): B=8, L=4096 (H=W=64), C=128, heads=4, hd=32
// window = vertical strip H_sp=64, W_sp=8 -> S=512 tokens per window,
// 64 windows x 4 heads = 256 attention problems of [512,32].
// Flash-attention with V_WMMA_F32_16X16X32_BF16 (fp32 accumulate).
// Softmax in exp2 domain; row reductions via fused VOP2+DPP16 arithmetic;
// P-matrix bf16 conversion by truncation (folds into ds_store_b16_d16_hi).
// ---------------------------------------------------------------------------

typedef __attribute__((ext_vector_type(16))) __bf16 v16bf;
typedef __attribute__((ext_vector_type(8)))  float  v8f;

#define BATCH     8
#define LTOK      4096
#define DIM       128
#define HEAD_DIM  32
#define SEQ       512
// qk scale folded together with log2(e): softmax(x) == softmax2(x*log2e)
#define SCALE_F   0.1767766952966369f          // 32^-0.5
#define LOG2E_F   1.4426950408889634f
#define QSCALE    (SCALE_F * LOG2E_F)

// LDS row strides (in u16) padded for bank-conflict-free reads
#define KSTR 34      // K rows: 68B = 17 dwords, coprime with 64 banks
#define VSTR 520     // V^T rows: 1040B = 260 dwords, 260%64=4 -> distinct banks
#define PSTR 34

#if __has_builtin(__builtin_amdgcn_exp2f)
#define EXP2F(x) __builtin_amdgcn_exp2f(x)     // bare v_exp_f32
#else
#define EXP2F(x) exp2f(x)
#endif

// round-to-nearest-even f32->bf16 (used for K/V/Q staging, off the hot path)
__device__ __forceinline__ unsigned short f2bf(float x) {
  unsigned int u = __float_as_uint(x);
  unsigned int r = u + 0x7FFFu + ((u >> 16) & 1u);
  return (unsigned short)(r >> 16);
}
// truncating f32->bf16 (hot path, P in [0,1]): pure high-half store, no ALU
__device__ __forceinline__ unsigned short f2bf_trunc(float x) {
  return (unsigned short)(__float_as_uint(x) >> 16);
}

union V16 {
  unsigned short u[16];
  v16bf v;
};

// Plain max as opaque asm: avoids llvm.maxnum's v_max(x,x) canonicalize chains
__device__ __forceinline__ float vmax(float a, float b) {
  float d;
  asm("v_max_num_f32 %0, %1, %2" : "=v"(d) : "v"(a), "v"(b));
  return d;
}

// 16-lane butterfly reductions as fused permute+op: one VOP2+DPP16 per step
// (v_max_num_f32_dpp d, x, x <perm> == d = max(perm(x), x)).
__device__ __forceinline__ float max16(float x) {
  float y;
  asm("v_max_num_f32_dpp %0, %1, %1 quad_perm:[1,0,3,2] row_mask:0xf bank_mask:0xf bound_ctrl:1"
      : "=v"(y) : "v"(x));
  asm("v_max_num_f32_dpp %0, %1, %1 quad_perm:[2,3,0,1] row_mask:0xf bank_mask:0xf bound_ctrl:1"
      : "=v"(x) : "v"(y));
  asm("v_max_num_f32_dpp %0, %1, %1 row_half_mirror row_mask:0xf bank_mask:0xf bound_ctrl:1"
      : "=v"(y) : "v"(x));
  asm("v_max_num_f32_dpp %0, %1, %1 row_mirror row_mask:0xf bank_mask:0xf bound_ctrl:1"
      : "=v"(x) : "v"(y));
  return x;
}
__device__ __forceinline__ float sum16(float x) {
  float y;
  asm("v_add_f32_dpp %0, %1, %1 quad_perm:[1,0,3,2] row_mask:0xf bank_mask:0xf bound_ctrl:1"
      : "=v"(y) : "v"(x));
  asm("v_add_f32_dpp %0, %1, %1 quad_perm:[2,3,0,1] row_mask:0xf bank_mask:0xf bound_ctrl:1"
      : "=v"(x) : "v"(y));
  asm("v_add_f32_dpp %0, %1, %1 row_half_mirror row_mask:0xf bank_mask:0xf bound_ctrl:1"
      : "=v"(y) : "v"(x));
  asm("v_add_f32_dpp %0, %1, %1 row_mirror row_mask:0xf bank_mask:0xf bound_ctrl:1"
      : "=v"(x) : "v"(y));
  return x;
}

__global__ __launch_bounds__(128) void
lepe_attn_fa_kernel(const float* __restrict__ qkv, float* __restrict__ out) {
  extern __shared__ unsigned short smem[];
  unsigned short* Ksh = smem;                       // [512][KSTR]  (key, d)
  unsigned short* Vsh = Ksh + SEQ * KSTR;           // [32][VSTR]   (d, key) transposed
  unsigned short* Psh = Vsh + HEAD_DIM * VSTR;      // [4 waves][16][PSTR]

  const int blk  = blockIdx.x;
  const int qg   = blk & 3;        // which quarter of the 32 Q tiles
  const int pair = blk >> 2;       // 0..255 = (window, head)
  const int hd   = pair & 3;
  const int win  = pair >> 2;      // 0..63
  const int b    = win >> 3;
  const int wi   = win & 7;        // vertical strip index

  const int tid  = threadIdx.x;
  const int lane = tid & 31;
  const int wave = tid >> 5;
  const int hi16 = lane >> 4;      // 0 | 1  (lane half)
  const int l16  = lane & 15;

  const size_t q_off = ((size_t)(0 * BATCH + b)) * LTOK * DIM;
  const size_t k_off = ((size_t)(1 * BATCH + b)) * LTOK * DIM;
  const size_t v_off = ((size_t)(2 * BATCH + b)) * LTOK * DIM;
  const int ch0 = hd * HEAD_DIM;

  // token s (0..511) in window -> row l in [0,4096): l = (s/8)*64 + wi*8 + (s%8)

  // ---- cooperative stage of K [512x32] and V^T [32x512] into LDS as bf16 ----
  {
    const float4* kg = (const float4*)(qkv + k_off);
    const float4* vg = (const float4*)(qkv + v_off);
#pragma unroll 4
    for (int i = 0; i < 32; ++i) {
      int idx = tid + (i << 7);          // 0..4095 : 512 rows x 8 float4
      int row = idx >> 3;                // key token s
      int d4  = (idx & 7) << 2;          // d offset 0,4,...,28
      int l   = ((row >> 3) << 6) + (wi << 3) + (row & 7);
      size_t g = (size_t)l * (DIM / 4) + (hd << 3) + (idx & 7);
      float4 kq = kg[g];
      float4 vq = vg[g];
      unsigned short* kr = Ksh + row * KSTR + d4;
      kr[0] = f2bf(kq.x); kr[1] = f2bf(kq.y); kr[2] = f2bf(kq.z); kr[3] = f2bf(kq.w);
      Vsh[(d4 + 0) * VSTR + row] = f2bf(vq.x);
      Vsh[(d4 + 1) * VSTR + row] = f2bf(vq.y);
      Vsh[(d4 + 2) * VSTR + row] = f2bf(vq.z);
      Vsh[(d4 + 3) * VSTR + row] = f2bf(vq.w);
    }
  }
  __syncthreads();

  unsigned short* Pw = Psh + wave * (16 * PSTR);

  // ---- each wave: two 16-row Q tiles, full flash-attention pass each ----
#pragma unroll
  for (int qt = 0; qt < 2; ++qt) {
    const int s0 = ((qg << 3) + (wave << 1) + qt) << 4;   // first Q row of tile

    // Q tile in WMMA A-layout (16x32 bf16), scaled by log2e/sqrt(d)
    V16 qa;
#pragma unroll
    for (int e = 0; e < 16; ++e) {
      int k = e + (hi16 ? 16 : 0);
      int s = s0 + l16;
      int l = ((s >> 3) << 6) + (wi << 3) + (s & 7);
      qa.u[e] = f2bf(qkv[q_off + (size_t)l * DIM + ch0 + k] * QSCALE);
    }

    v8f o0 = {};
    v8f o1 = {};
    float mrun[8], lrun[8];   // lrun: per-lane PARTIAL row sum (reduced once at end)
#pragma unroll
    for (int r = 0; r < 8; ++r) { mrun[r] = -__builtin_inff(); lrun[r] = 0.0f; }

#pragma unroll 2
    for (int ic = 0; ic < 16; ++ic) {     // 32 keys per chunk
      const int kb = ic << 5;

      // K^T as two B-tiles: N = key (l16), reduction K = head-dim
      V16 kb0, kb1;
#pragma unroll
      for (int e = 0; e < 16; ++e) {
        int d = e + (hi16 ? 16 : 0);
        kb0.u[e] = Ksh[(kb + l16) * KSTR + d];
        kb1.u[e] = Ksh[(kb + 16 + l16) * KSTR + d];
      }

      v8f cz = {};
      v8f S0 = __builtin_amdgcn_wmma_f32_16x16x32_bf16(
          false, qa.v, false, kb0.v, (short)0, cz, false, false);
      v8f S1 = __builtin_amdgcn_wmma_f32_16x16x32_bf16(
          false, qa.v, false, kb1.v, (short)0, cz, false, false);

      // online softmax (exp2 domain): rows r / r+8 per lane-half
      float p0[8], p1[8];
#pragma unroll
      for (int r = 0; r < 8; ++r) {
        float rm    = max16(vmax(S0[r], S1[r]));   // fused DPP butterfly
        float mn    = vmax(mrun[r], rm);
        float alpha = EXP2F(mrun[r] - mn);         // bare v_exp_f32
        mrun[r] = mn;
        float e0 = EXP2F(S0[r] - mn);
        float e1 = EXP2F(S1[r] - mn);
        p0[r] = e0; p1[r] = e1;
        // per-lane partial denominator: alpha is lane-uniform per row, so the
        // 16-lane sum reduction can be deferred to after the kv loop
        lrun[r] = lrun[r] * alpha + e0 + e1;
        o0[r] *= alpha;
        o1[r] *= alpha;
      }

      // accumulator layout -> A layout via per-wave LDS scratch (same-wave
      // DS ops are in-order: no barrier needed). Truncating bf16: the >>16
      // folds into ds_store_b16_d16_hi, zero ALU cost.
#pragma unroll
      for (int r = 0; r < 8; ++r) {
        int m = r + (hi16 << 3);
        Pw[m * PSTR + l16]      = f2bf_trunc(p0[r]);
        Pw[m * PSTR + 16 + l16] = f2bf_trunc(p1[r]);
      }
      V16 pa;
#pragma unroll
      for (int e = 0; e < 16; ++e) {
        int k = e + (hi16 ? 16 : 0);
        pa.u[e] = Pw[l16 * PSTR + k];
      }

      // V chunk as two B-tiles: N = head-dim (l16), reduction K = key
      V16 vb0, vb1;
#pragma unroll
      for (int e = 0; e < 16; ++e) {
        int key = kb + e + (hi16 ? 16 : 0);
        vb0.u[e] = Vsh[l16 * VSTR + key];
        vb1.u[e] = Vsh[(16 + l16) * VSTR + key];
      }

      o0 = __builtin_amdgcn_wmma_f32_16x16x32_bf16(
          false, pa.v, false, vb0.v, (short)0, o0, false, false);
      o1 = __builtin_amdgcn_wmma_f32_16x16x32_bf16(
          false, pa.v, false, vb1.v, (short)0, o1, false, false);
    }

    // normalize (single deferred 16-lane sum reduction per row) and write:
    // rows r+8*hi16, cols = lane across 16
#pragma unroll
    for (int r = 0; r < 8; ++r) {
      float inv = 1.0f / sum16(lrun[r]);
      int s = s0 + r + (hi16 << 3);
      int l = ((s >> 3) << 6) + (wi << 3) + (s & 7);
      size_t base = ((size_t)b * LTOK + l) * DIM + ch0;
      out[base + l16]      = o0[r] * inv;
      out[base + 16 + l16] = o1[r] * inv;
    }
  }
}

extern "C" void kernel_launch(void* const* d_in, const int* in_sizes, int n_in,
                              void* d_out, int out_size, void* d_ws, size_t ws_size,
                              hipStream_t stream) {
  const float* qkv = (const float*)d_in[0];   // [3,8,4096,128] f32 (H=W=64 fixed)
  float* out = (float*)d_out;                 // [8,4096,128] f32

  // 256 (window,head) pairs x 4 q-groups; 128 threads = 4 waves of wave32
  dim3 grid(1024), block(128);
  size_t shmem = (size_t)(SEQ * KSTR + HEAD_DIM * VSTR + 4 * 16 * PSTR)
                 * sizeof(unsigned short);    // ~72 KB, fits 320 KB WGP LDS
  lepe_attn_fa_kernel<<<grid, block, shmem, stream>>>(qkv, out);
}
